// model_3917010174081
// MI455X (gfx1250) — compile-verified, matrix operand order checked by reading, not run on hardware
//
#include <hip/hip_runtime.h>
#include <math.h>

// ---------------------------------------------------------------------------
// Types for CDNA5 WMMA (wave32, v_wmma_f32_16x16x32_bf16)
// ---------------------------------------------------------------------------
typedef __bf16 bf16_t;
typedef __attribute__((ext_vector_type(16))) __bf16 v16bf;
typedef __attribute__((ext_vector_type(8)))  __bf16 v8bf;
typedef __attribute__((ext_vector_type(8)))  float  v8f;

// ---------------------------------------------------------------------------
// Implicit-GEMM conv + fused BN (+residual) (+ReLU), bf16 in / f32 acc / bf16 out
//   GEMM view: M = Cout, N = B*OH*OW, K = C*R*S
//   Workgroup: 128 threads (4 waves), 64x64 output tile, wave = 32x32 (2x2 WMMA)
// ---------------------------------------------------------------------------
struct ConvP {
  const bf16_t* x;      // [B,C,IH,IW]
  const bf16_t* w;      // [M,Kpad] (bf16, zero padded Kdim->Kpad)
  const float*  bn_s; const float* bn_b; const float* bn_m; const float* bn_v;
  const bf16_t* res;    // residual [B,M,OH,OW] or nullptr
  bf16_t*       y;      // [B,M,OH,OW]
  int C, IH, IW, M, OH, OW, R, S, stride, pad, Kdim, Kpad, Ntot, relu;
};

#define LDSTRIDE 40   // 32 K-elems + 8 pad (bf16) per row: 16B-aligned, fewer bank conflicts

__global__ __launch_bounds__(128) void conv_bn_wmma(ConvP p) {
  __shared__ bf16_t Asb[64][LDSTRIDE];   // A tile: [m_local][k_local]  (weights)
  __shared__ bf16_t Bsb[64][LDSTRIDE];   // B tile: [n_local][k_local]  (im2col gather)

  const int tid  = threadIdx.x;
  const int lane = tid & 31;
  const int wave = tid >> 5;
  const int wm   = (wave >> 1) * 32;     // wave m-offset in tile (0/32)
  const int wn   = (wave & 1)  * 32;     // wave n-offset in tile (0/32)
  const int bm   = blockIdx.y * 64;
  const int bn   = blockIdx.x * 64;

  // Cooperative-load assignment: each thread owns one 64-row's 16-elem K segment
  const int row  = tid >> 1;             // 0..63
  const int kseg = (tid & 1) * 16;       // 0 or 16

  // Decompose this thread's B-tile column n -> (b, oy, ox) once (fixed over K loop)
  const int ng = bn + row;
  const bool nvalid = (ng < p.Ntot);
  int b_i, oy, ox;
  {
    int t = nvalid ? ng : 0;
    const int ohw = p.OH * p.OW;
    b_i = t / ohw; int rm = t - b_i * ohw;
    oy = rm / p.OW; ox = rm - oy * p.OW;
  }
  // Incremental (c,r,s) decomposition of this thread's K base (kk0 + kseg)
  int c0, r0, s0;
  { int k = kseg; const int RS = p.R * p.S;
    c0 = k / RS; int rr = k - c0 * RS; r0 = rr / p.S; s0 = rr - r0 * p.S; }

  v8f acc[2][2] = {};

  for (int kk0 = 0; kk0 < p.Kdim; kk0 += 32) {
    // ---- A tile: weights (pre-padded, vector loads) ----
    {
      const int mg = bm + row;
      v8bf z = {};
      v8bf w0 = z, w1 = z;
      if (mg < p.M) {
        const bf16_t* src = p.w + (size_t)mg * p.Kpad + kk0 + kseg;
        w0 = *(const v8bf*)(src);
        w1 = *(const v8bf*)(src + 8);
        if (kk0 + 32 < p.Kdim)
          __builtin_prefetch(src + 32, 0, 1);   // gfx1250 global_prefetch_b8
      }
      *(v8bf*)&Asb[row][kseg]     = w0;
      *(v8bf*)&Asb[row][kseg + 8] = w1;
    }
    // ---- B tile: im2col gather into [n_local][k_local] ----
    {
      int c = c0, r = r0, s = s0;
#pragma unroll
      for (int e = 0; e < 16; ++e) {
        bf16_t v = (bf16_t)0.f;
        if (nvalid && (kk0 + kseg + e) < p.Kdim) {
          const int iy = oy * p.stride - p.pad + r;
          const int ix = ox * p.stride - p.pad + s;
          if ((unsigned)iy < (unsigned)p.IH && (unsigned)ix < (unsigned)p.IW)
            v = p.x[(((size_t)b_i * p.C + c) * p.IH + iy) * p.IW + ix];
        }
        Bsb[row][kseg + e] = v;
        if (++s == p.S) { s = 0; if (++r == p.R) { r = 0; ++c; } }
      }
    }
    __syncthreads();
    // advance base (c0,r0,s0) by 32 for next iteration
    { s0 += 32;
      while (s0 >= p.S) { s0 -= p.S; if (++r0 == p.R) { r0 = 0; ++c0; } } }

    // ---- fragments per documented 16-bit A/B layouts ----
    const int fr = lane & 15;
    const int kb = (lane >> 4) * 8;      // lanes 0-15: K 0..7/16..23 ; lanes 16-31: K 8..15/24..31
    v16bf afrag[2], bfrag[2];
#pragma unroll
    for (int i = 0; i < 2; ++i) {
      v8bf alo = *(const v8bf*)&Asb[wm + i * 16 + fr][kb];
      v8bf ahi = *(const v8bf*)&Asb[wm + i * 16 + fr][kb + 16];
      afrag[i] = __builtin_shufflevector(alo, ahi, 0,1,2,3,4,5,6,7,8,9,10,11,12,13,14,15);
      v8bf blo = *(const v8bf*)&Bsb[wn + i * 16 + fr][kb];
      v8bf bhi = *(const v8bf*)&Bsb[wn + i * 16 + fr][kb + 16];
      bfrag[i] = __builtin_shufflevector(blo, bhi, 0,1,2,3,4,5,6,7,8,9,10,11,12,13,14,15);
    }
#pragma unroll
    for (int i = 0; i < 2; ++i)
#pragma unroll
      for (int j = 0; j < 2; ++j)
        acc[i][j] = __builtin_amdgcn_wmma_f32_16x16x32_bf16(
            false, afrag[i], false, bfrag[j], (short)0, acc[i][j], false, false);
    __syncthreads();
  }

  // ---- epilogue: fused BN (+ residual) (+ ReLU), bf16 store to NCHW ----
  const int fr    = lane & 15;
  const int mhalf = (lane >> 4) * 8;     // C/D layout: lanes 16-31 hold rows M+8..M+15
  const int ohw   = p.OH * p.OW;
#pragma unroll
  for (int j = 0; j < 2; ++j) {
    const int n_gl = bn + wn + j * 16 + fr;
    if (n_gl >= p.Ntot) continue;
    const int bb = n_gl / ohw; int rm = n_gl - bb * ohw;
    const int yy = rm / p.OW; const int xx = rm - yy * p.OW;
#pragma unroll
    for (int i = 0; i < 2; ++i) {
#pragma unroll
      for (int rI = 0; rI < 8; ++rI) {
        const int m_gl = bm + wm + i * 16 + mhalf + rI;
        if (m_gl >= p.M) continue;
        const float inv = p.bn_s[m_gl] * rsqrtf(p.bn_v[m_gl] + 1e-5f);
        float val = acc[i][j][rI] * inv + (p.bn_b[m_gl] - p.bn_m[m_gl] * inv);
        const size_t idx = (((size_t)bb * p.M + m_gl) * p.OH + yy) * p.OW + xx;
        if (p.res) val += (float)p.res[idx];
        if (p.relu) val = val > 0.f ? val : 0.f;
        p.y[idx] = (bf16_t)val;
      }
    }
  }
}

// ---------------------------------------------------------------------------
// Small helper kernels
// ---------------------------------------------------------------------------
__global__ void k_cvt_f32_bf16(const float* s, bf16_t* d, int n) {
  int i = blockIdx.x * 256 + threadIdx.x;
  if (i < n) d[i] = (bf16_t)s[i];
}

__global__ void k_cvt_w(const float* src, bf16_t* dst, int M, int Kdim, int Kpad) {
  int i = blockIdx.x * 256 + threadIdx.x;
  if (i >= M * Kpad) return;
  int m = i / Kpad, k = i - m * Kpad;
  dst[i] = (bf16_t)((k < Kdim) ? src[(size_t)m * Kdim + k] : 0.f);
}

__global__ void k_maxpool(const bf16_t* x, bf16_t* y, int BC, int IH, int IW, int OH, int OW) {
  int i = blockIdx.x * 256 + threadIdx.x;
  if (i >= BC * OH * OW) return;
  int bc = i / (OH * OW); int r = i - bc * OH * OW;
  int oy = r / OW, ox = r - oy * OW;
  float m = -3.4e38f;
  for (int ky = 0; ky < 3; ++ky)
    for (int kx = 0; kx < 3; ++kx) {
      int iy = oy * 2 - 1 + ky, ix = ox * 2 - 1 + kx;
      if ((unsigned)iy < (unsigned)IH && (unsigned)ix < (unsigned)IW)
        m = fmaxf(m, (float)x[((size_t)bc * IH + iy) * IW + ix]);
    }
  y[i] = (bf16_t)m;
}

__global__ void k_avgpool(const bf16_t* x, float* f, int BC, int HW) {
  int i = blockIdx.x * 256 + threadIdx.x;
  if (i >= BC) return;
  float s = 0.f;
  for (int j = 0; j < HW; ++j) s += (float)x[(size_t)i * HW + j];
  f[i] = s / (float)HW;
}

// ---------------------------------------------------------------------------
// Tail: char-linear -> 2x GRU -> attention -> LN1   (single workgroup; whole
// network collapses to batch-1 seq-32 dim-94 because x is broadcast over n)
// ---------------------------------------------------------------------------
__device__ __forceinline__ float sigm(float x) { return 1.f / (1.f + expf(-x)); }

__device__ void gru_layer(const float* IN, float* OUT,
                          const float* wih, const float* whh,
                          const float* bih, const float* bhh,
                          float* Hh, float* XW, float* HW, int tid) {
  for (int d = tid; d < 94; d += 256) Hh[d] = 0.f;
  __syncthreads();
  for (int t = 0; t < 32; ++t) {
    for (int g = tid; g < 282; g += 256) {
      float sx = bih[g], sh = bhh[g];
      const float* xr = IN + t * 94;
      const float* wi = wih + g * 94;
      const float* wh = whh + g * 94;
      for (int k = 0; k < 94; ++k) { sx += xr[k] * wi[k]; sh += Hh[k] * wh[k]; }
      XW[g] = sx; HW[g] = sh;
    }
    __syncthreads();
    if (tid < 94) {
      const int d = tid;
      float r = sigm(XW[d] + HW[d]);
      float z = sigm(XW[94 + d] + HW[94 + d]);
      float n = tanhf(XW[188 + d] + r * HW[188 + d]);
      float hn = (1.f - z) * n + z * Hh[d];
      OUT[t * 94 + d] = hn; Hh[d] = hn;
    }
    __syncthreads();
  }
}

struct TailP {
  const float* f;  const float* cw; const float* cb;
  const float* wih0; const float* whh0; const float* bih0; const float* bhh0;
  const float* wih1; const float* whh1; const float* bih1; const float* bhh1;
  const float* wq; const float* bq; const float* wk; const float* bk;
  const float* wv; const float* bv; const float* wo; const float* bo;
  const float* ln1g; const float* ln1b;
  float* xln;
};

__global__ __launch_bounds__(256) void k_tail1(TailP p) {
  __shared__ float S0[3008], S1[3008], S2[3008], S3[3008];  // 32*94 each
  __shared__ float SC[1024];                                 // 32*32 scores
  __shared__ float Hh[94], XW[282], HW[282];
  const int tid = threadIdx.x;

  // c = relu(f @ char_w.T + char_b)  -> S0
  for (int i = tid; i < 3008; i += 256) {
    int t = i / 94, d = i - t * 94;
    float s = p.cb[d];
    const float* fr = p.f + t * 512;
    const float* wr = p.cw + d * 512;
    for (int k = 0; k < 512; ++k) s += fr[k] * wr[k];
    S0[i] = s > 0.f ? s : 0.f;
  }
  __syncthreads();

  gru_layer(S0, S1, p.wih0, p.whh0, p.bih0, p.bhh0, Hh, XW, HW, tid);  // layer 1
  gru_layer(S1, S0, p.wih1, p.whh1, p.bih1, p.bhh1, Hh, XW, HW, tid);  // layer 2 -> S0

  // q,k,v
  for (int i = tid; i < 3008; i += 256) {
    int t = i / 94, d = i - t * 94;
    float sq = p.bq[d], sk = p.bk[d], sv = p.bv[d];
    const float* xr = S0 + t * 94;
    for (int k = 0; k < 94; ++k) {
      float xv = xr[k];
      sq += xv * p.wq[d * 94 + k];
      sk += xv * p.wk[d * 94 + k];
      sv += xv * p.wv[d * 94 + k];
    }
    S1[i] = sq; S2[i] = sk; S3[i] = sv;
  }
  __syncthreads();

  const float isq = 0.1031421250f;  // 1/sqrt(94)
  for (int i = tid; i < 1024; i += 256) {
    int t = i >> 5, s2 = i & 31;
    float a = 0.f;
    for (int k = 0; k < 94; ++k) a += S1[t * 94 + k] * S2[s2 * 94 + k];
    SC[i] = a * isq;
  }
  __syncthreads();
  if (tid < 32) {
    float mx = -3.4e38f;
    for (int s2 = 0; s2 < 32; ++s2) mx = fmaxf(mx, SC[tid * 32 + s2]);
    float sum = 0.f;
    for (int s2 = 0; s2 < 32; ++s2) { float e = expf(SC[tid * 32 + s2] - mx); SC[tid * 32 + s2] = e; sum += e; }
    float inv = 1.f / sum;
    for (int s2 = 0; s2 < 32; ++s2) SC[tid * 32 + s2] *= inv;
  }
  __syncthreads();
  // av -> S1 (q dead)
  for (int i = tid; i < 3008; i += 256) {
    int t = i / 94, d = i - t * 94;
    float a = 0.f;
    for (int s2 = 0; s2 < 32; ++s2) a += SC[t * 32 + s2] * S3[s2 * 94 + d];
    S1[i] = a;
  }
  __syncthreads();
  // x + av@wo.T + bo  -> S2 (k dead)
  for (int i = tid; i < 3008; i += 256) {
    int t = i / 94, d = i - t * 94;
    float a = p.bo[d];
    for (int k = 0; k < 94; ++k) a += S1[t * 94 + k] * p.wo[d * 94 + k];
    S2[i] = S0[i] + a;
  }
  __syncthreads();
  // LN1 -> global xln
  if (tid < 32) {
    const int t = tid;
    float mu = 0.f;
    for (int d = 0; d < 94; ++d) mu += S2[t * 94 + d];
    mu /= 94.f;
    float var = 0.f;
    for (int d = 0; d < 94; ++d) { float df = S2[t * 94 + d] - mu; var += df * df; }
    var /= 94.f;
    float inv = rsqrtf(var + 1e-5f);
    for (int d = 0; d < 94; ++d)
      p.xln[t * 94 + d] = (S2[t * 94 + d] - mu) * inv * p.ln1g[d] + p.ln1b[d];
  }
}

__global__ void k_ff1(const float* xln, const float* w1, const float* b1, float* ffh) {
  int i = blockIdx.x * 256 + threadIdx.x;
  if (i >= 32 * 2048) return;
  int t = i >> 11, j = i & 2047;
  float s = b1[j];
  const float* xr = xln + t * 94;
  const float* wr = w1 + j * 94;
  for (int k = 0; k < 94; ++k) s += xr[k] * wr[k];
  ffh[i] = s > 0.f ? s : 0.f;
}

__global__ void k_ff2(const float* xln, const float* ffh, const float* w2, const float* b2, float* y) {
  int t = blockIdx.x, d = threadIdx.x;
  if (d >= 94) return;
  float s = b2[d];
  const float* hr = ffh + t * 2048;
  const float* wr = w2 + d * 2048;
  for (int k = 0; k < 2048; ++k) s += hr[k] * wr[k];
  y[t * 94 + d] = xln[t * 94 + d] + s;
}

__global__ void k_ln2_softmax(const float* y, const float* g, const float* b, float* sd) {
  __shared__ float tmp[3008];
  int t = threadIdx.x;
  if (t < 32) {
    float mu = 0.f;
    for (int d = 0; d < 94; ++d) mu += y[t * 94 + d];
    mu /= 94.f;
    float var = 0.f;
    for (int d = 0; d < 94; ++d) { float df = y[t * 94 + d] - mu; var += df * df; }
    var /= 94.f;
    float inv = rsqrtf(var + 1e-5f);
    float mx = -3.4e38f;
    for (int d = 0; d < 94; ++d) {
      float v = (y[t * 94 + d] - mu) * inv * g[d] + b[d];
      tmp[t * 94 + d] = v; mx = fmaxf(mx, v);
    }
    float sum = 0.f;
    for (int d = 0; d < 94; ++d) { float e = expf(tmp[t * 94 + d] - mx); tmp[t * 94 + d] = e; sum += e; }
    float is = 1.f / sum;
    for (int d = 0; d < 94; ++d) sd[t * 94 + d] = tmp[t * 94 + d] * is;
  }
}

__global__ void k_bcast(const float* sd, float* out) {
  int s = blockIdx.x, b = blockIdx.y, d = threadIdx.x;
  if (d < 94) out[((size_t)(b * 512 + s)) * 94 + d] = sd[b * 94 + d];
}

// ---------------------------------------------------------------------------
// Host orchestration
// ---------------------------------------------------------------------------
struct WB { bf16_t* w; int Kdim; int Kpad; };

static void launch_conv(hipStream_t st, const bf16_t* x, const WB& w,
                        const float* s, const float* b, const float* m, const float* v,
                        const bf16_t* res, bf16_t* y,
                        int B, int C, int IH, int IW, int M, int R, int S,
                        int stride, int pad, int relu) {
  ConvP p;
  p.x = x; p.w = w.w; p.bn_s = s; p.bn_b = b; p.bn_m = m; p.bn_v = v;
  p.res = res; p.y = y;
  p.C = C; p.IH = IH; p.IW = IW; p.M = M;
  p.OH = (IH + 2 * pad - R) / stride + 1;
  p.OW = (IW + 2 * pad - S) / stride + 1;
  p.R = R; p.S = S; p.stride = stride; p.pad = pad;
  p.Kdim = w.Kdim; p.Kpad = w.Kpad;
  p.Ntot = B * p.OH * p.OW; p.relu = relu;
  dim3 grid((p.Ntot + 63) / 64, (M + 63) / 64);
  conv_bn_wmma<<<grid, dim3(128), 0, st>>>(p);
}

extern "C" void kernel_launch(void* const* d_in, const int* in_sizes, int n_in,
                              void* d_out, int out_size, void* d_ws, size_t ws_size,
                              hipStream_t stream) {
  (void)in_sizes; (void)n_in; (void)out_size; (void)ws_size;
  auto F = [&](int i) { return (const float*)d_in[i]; };

  // bump allocator over workspace
  char* base = (char*)d_ws;
  size_t off = 0;
  auto alloc = [&](size_t elems, size_t esz) -> void* {
    off = (off + 255) & ~(size_t)255;
    void* p = base + off;
    off += elems * esz;
    return p;
  };

  bf16_t* xbf  = (bf16_t*)alloc(32ull * 3 * 224 * 224, 2);
  bf16_t* c1o  = (bf16_t*)alloc(32ull * 64 * 112 * 112, 2);
  const size_t PE = 32ull * 64 * 56 * 56;  // largest post-pool activation
  bf16_t* P[3] = { (bf16_t*)alloc(PE, 2), (bf16_t*)alloc(PE, 2), (bf16_t*)alloc(PE, 2) };
  bf16_t* Dbuf = (bf16_t*)alloc(PE, 2);
  float* fvec  = (float*)alloc(32 * 512, 4);
  float* xln   = (float*)alloc(32 * 94, 4);
  float* ffh   = (float*)alloc(32 * 2048, 4);
  float* ybuf  = (float*)alloc(32 * 94, 4);
  float* sden  = (float*)alloc(32 * 94, 4);

  auto cvt_w = [&](int widx, int M, int C, int R, int S) -> WB {
    WB wb; wb.Kdim = C * R * S; wb.Kpad = (wb.Kdim + 31) & ~31;
    wb.w = (bf16_t*)alloc((size_t)M * wb.Kpad, 2);
    int n = M * wb.Kpad;
    k_cvt_w<<<(n + 255) / 256, 256, 0, stream>>>(F(widx), wb.w, M, wb.Kdim, wb.Kpad);
    return wb;
  };

  // ---- stem: convert input, conv1 7x7/2, maxpool 3x3/2 ----
  { int n = 32 * 3 * 224 * 224;
    k_cvt_f32_bf16<<<(n + 255) / 256, 256, 0, stream>>>(F(0), xbf, n); }
  WB w0 = cvt_w(1, 64, 3, 7, 7);
  launch_conv(stream, xbf, w0, F(2), F(3), F(4), F(5), nullptr, c1o,
              32, 3, 224, 224, 64, 7, 7, 2, 3, 1);
  { int n = 32 * 64 * 56 * 56;
    k_maxpool<<<(n + 255) / 256, 256, 0, stream>>>(c1o, P[0], 32 * 64, 112, 112, 56, 56); }

  // ---- residual blocks (d_in indices per setup_inputs insertion order) ----
  struct BC { int w1, b1, w2, b2, wd, bd, cout; };
  const BC blocks[8] = {
    { 6,  7, 11, 12, -1, -1,  64}, {16, 17, 21, 22, -1, -1,  64},
    {26, 27, 31, 32, 36, 37, 128}, {41, 42, 46, 47, -1, -1, 128},
    {51, 52, 56, 57, 61, 62, 256}, {66, 67, 71, 72, -1, -1, 256},
    {76, 77, 81, 82, 86, 87, 512}, {91, 92, 96, 97, -1, -1, 512},
  };
  int H = 56, C = 64, cur = 0;
  for (int bi = 0; bi < 8; ++bi) {
    const BC& bc = blocks[bi];
    const int stride = (bc.wd >= 0) ? 2 : 1;
    const int OH = (stride == 2) ? H / 2 : H;
    const int t = (cur + 1) % 3, o = (cur + 2) % 3;

    WB w1 = cvt_w(bc.w1, bc.cout, C, 3, 3);
    launch_conv(stream, P[cur], w1, F(bc.b1), F(bc.b1 + 1), F(bc.b1 + 2), F(bc.b1 + 3),
                nullptr, P[t], 32, C, H, H, bc.cout, 3, 3, stride, 1, 1);

    const bf16_t* ident = P[cur];
    if (bc.wd >= 0) {
      WB wd = cvt_w(bc.wd, bc.cout, C, 1, 1);
      launch_conv(stream, P[cur], wd, F(bc.bd), F(bc.bd + 1), F(bc.bd + 2), F(bc.bd + 3),
                  nullptr, Dbuf, 32, C, H, H, bc.cout, 1, 1, stride, 0, 0);
      ident = Dbuf;
    }

    WB w2 = cvt_w(bc.w2, bc.cout, bc.cout, 3, 3);
    launch_conv(stream, P[t], w2, F(bc.b2), F(bc.b2 + 1), F(bc.b2 + 2), F(bc.b2 + 3),
                ident, P[o], 32, bc.cout, OH, OH, bc.cout, 3, 3, 1, 1, 1);

    cur = o; H = OH; C = bc.cout;
  }

  // ---- global average pool -> f32 [32,512] ----
  k_avgpool<<<(16384 + 255) / 256, 256, 0, stream>>>(P[cur], fvec, 32 * 512, 49);

  // ---- collapsed tail (n-broadcast structure): char-linear, GRUx2, attn, LN1 ----
  TailP tp = { fvec, F(125), F(126),
               F(101), F(102), F(103), F(104),
               F(105), F(106), F(107), F(108),
               F(109), F(110), F(111), F(112), F(113), F(114), F(115), F(116),
               F(121), F(122), xln };
  k_tail1<<<1, 256, 0, stream>>>(tp);
  k_ff1<<<256, 256, 0, stream>>>(xln, F(117), F(118), ffh);
  k_ff2<<<32, 128, 0, stream>>>(xln, ffh, F(119), F(120), ybuf);
  k_ln2_softmax<<<1, 32, 0, stream>>>(ybuf, F(123), F(124), sden);
  k_bcast<<<dim3(512, 32), 96, 0, stream>>>(sden, (float*)d_out);
}
